// SelfMatchingAttention_83408264888682
// MI455X (gfx1250) — compile-verified, matrix operand order checked by reading, not run on hardware
//
#include <hip/hip_runtime.h>
#include <hip/hip_bf16.h>
#include <math.h>

typedef __attribute__((ext_vector_type(16))) _Float16 v16h;
typedef __attribute__((ext_vector_type(8)))  _Float16 v8h;
typedef __attribute__((ext_vector_type(8)))  float    v8f;

#define LSEQ 512
#define NB   4      // batch B
#define DD   256    // D
#define HH   128    // H

// ---------------------------------------------------------------------------
// f32 -> f16 convert
// ---------------------------------------------------------------------------
__global__ void f32_to_f16_kernel(const float* __restrict__ src,
                                  _Float16* __restrict__ dst, int n) {
  int i = blockIdx.x * blockDim.x + threadIdx.x;
  if (i < n) dst[i] = (_Float16)src[i];
}

// ---------------------------------------------------------------------------
// WMMA fragment load: 16x32 f16 tile, row-major source with leading dim `ld`.
// lane&15 selects the row (M for A-frag, N for B-frag fed as W[n,k]);
// K striping per CDNA5 ISA 16-bit layout: {k0+8h .. +7, k0+16+8h .. +7}.
// ---------------------------------------------------------------------------
__device__ __forceinline__ v16h load_frag(const _Float16* __restrict__ p,
                                          int ld, int row0, int k0) {
  const int lane = threadIdx.x & 31;
  const int r    = row0 + (lane & 15);
  const int half = lane >> 4;
  const _Float16* base = p + (size_t)r * ld + (k0 + 8 * half);
  v8h lo = *(const v8h*)(base);
  v8h hi = *(const v8h*)(base + 16);
  v16h out;
#pragma unroll
  for (int i = 0; i < 8; ++i) { out[i] = lo[i]; out[i + 8] = hi[i]; }
  return out;
}

// ---------------------------------------------------------------------------
// WMMA GEMM: C[M,N] = A[M,K] * W[N,K]^T (+ epilogue)
//   EP=0: C_f32 = acc
//   EP=1: C_f32 = acc + bias[n]
//   EP=2: Ch_f16 = gsrc[m,n] * sigmoid(acc)     (gated activation, fused)
// One 16x16 C-tile per wave; 4 waves (128 threads) per block.
// ---------------------------------------------------------------------------
template <int EP>
__global__ __launch_bounds__(128)
void wmma_gemm_kernel(const _Float16* __restrict__ A,
                      const _Float16* __restrict__ W,
                      float* __restrict__ C,
                      const float* __restrict__ bias,
                      const float* __restrict__ gsrc,
                      _Float16* __restrict__ Ch,
                      int M, int N, int K) {
  const int lane = threadIdx.x & 31;
  const int wv   = threadIdx.x >> 5;
  const int ntn  = N >> 4;
  const int tile = blockIdx.x * 4 + wv;
  const int mt = tile / ntn, nt = tile % ntn;
  const int m0 = mt << 4, n0 = nt << 4;

  v8f acc = {};
#pragma unroll 4
  for (int k0 = 0; k0 < K; k0 += 32) {
    v16h af = load_frag(A, K, m0, k0);
    v16h bf = load_frag(W, K, n0, k0);
    acc = __builtin_amdgcn_wmma_f32_16x16x32_f16(
        /*neg_a=*/false, af, /*neg_b=*/false, bf,
        /*c_mod=*/(short)0, acc, /*reuse_a=*/false, /*reuse_b=*/false);
  }

  const int nn = n0 + (lane & 15);
  const int mh = 8 * (lane >> 4);
#pragma unroll
  for (int r = 0; r < 8; ++r) {
    const int row = m0 + mh + r;
    const size_t idx = (size_t)row * N + nn;
    float v = acc[r];
    if (EP == 1) v += bias[nn];
    if (EP == 2) {
      const float g = 1.0f / (1.0f + __expf(-v));
      Ch[idx] = (_Float16)(gsrc[idx] * g);
    } else {
      C[idx] = v;
    }
  }
}

// ---------------------------------------------------------------------------
// Additive attention + softmax + context, one block per (j, b).
//   sj[i] = sum_h v[h]*tanh(a[i,b,h] + bj[h]); softmax over i;
//   ct[d] = sum_i w[i]*passage[i,b,d];  emits cat=[passage|ct] in f32 and f16.
// ---------------------------------------------------------------------------
__global__ __launch_bounds__(256)
void attention_kernel(const float* __restrict__ passage,
                      const float* __restrict__ mask,
                      const float* __restrict__ vT,
                      const float* __restrict__ abuf,
                      const float* __restrict__ bbuf,
                      float* __restrict__ cat32,
                      _Float16* __restrict__ cat16) {
  __shared__ float bj[HH], vv[HH], sj[LSEQ], red[256];
  const int blk  = blockIdx.x;        // = j*NB + b
  const int bIdx = blk % NB;
  const int tid  = threadIdx.x;

  if (tid < HH) {
    bj[tid] = bbuf[(size_t)blk * HH + tid];
    vv[tid] = vT[tid];
  }
  __syncthreads();

  const int lane = tid & 31, wv = tid >> 5;
  for (int i = wv; i < LSEQ; i += 8) {
    const float* arow = abuf + ((size_t)i * NB + bIdx) * HH;
    float p = 0.f;
#pragma unroll
    for (int s = 0; s < 4; ++s) {
      const int h = lane + 32 * s;
      p += vv[h] * tanhf(arow[h] + bj[h]);
    }
#pragma unroll
    for (int off = 16; off > 0; off >>= 1) p += __shfl_xor(p, off, 32);
    if (lane == 0) {
      const float mk = mask[(size_t)i * NB + bIdx];
      sj[i] = (mk > 0.f) ? p : -1e30f;
    }
  }
  __syncthreads();

  // block max
  float m = -1e30f;
  for (int i = tid; i < LSEQ; i += 256) m = fmaxf(m, sj[i]);
  red[tid] = m; __syncthreads();
  for (int s = 128; s > 0; s >>= 1) {
    if (tid < s) red[tid] = fmaxf(red[tid], red[tid + s]);
    __syncthreads();
  }
  const float mx = red[0];
  __syncthreads();

  // block sum of exp
  float sum = 0.f;
  for (int i = tid; i < LSEQ; i += 256) sum += __expf(sj[i] - mx);
  red[tid] = sum; __syncthreads();
  for (int s = 128; s > 0; s >>= 1) {
    if (tid < s) red[tid] += red[tid + s];
    __syncthreads();
  }
  const float inv = 1.0f / red[0];
  __syncthreads();

  for (int i = tid; i < LSEQ; i += 256) sj[i] = __expf(sj[i] - mx) * inv;
  __syncthreads();

  // context + concat (256 threads == DD)
  const int d = tid;
  float acc = 0.f;
  for (int i = 0; i < LSEQ; ++i)
    acc += sj[i] * passage[((size_t)i * NB + bIdx) * DD + d];

  const float pv = passage[(size_t)blk * DD + d];
  const size_t r0 = (size_t)blk * (2 * DD);
  cat32[r0 + d]       = pv;   cat16[r0 + d]       = (_Float16)pv;
  cat32[r0 + DD + d]  = acc;  cat16[r0 + DD + d]  = (_Float16)acc;
}

// ---------------------------------------------------------------------------
// Bidirectional GRU scan for one layer. blockIdx.x = direction.
// Persistent: whh^T cached in 196KB dynamic LDS (of the 320KB WGP pool),
// h state in LDS. gi (= x@wih.T + bih) precomputed by the WMMA kernel.
// 512 threads: phase A computes gh = h@whh.T + bhh (3 dots/thread),
// phase B computes one (b, j) GRU cell update per thread.
// ---------------------------------------------------------------------------
__global__ __launch_bounds__(512)
void gru_scan_kernel(const float* __restrict__ gi0, const float* __restrict__ gi1,
                     const float* __restrict__ whh0, const float* __restrict__ whh1,
                     const float* __restrict__ bhh0, const float* __restrict__ bhh1,
                     _Float16* __restrict__ xh16,   // (L*NB, 2*HH) next-layer input
                     float* __restrict__ out32) {   // nullable, final output
  extern __shared__ float whhT[];                   // [HH][384]
  __shared__ float hcur[NB * HH];                   // 512 floats
  __shared__ float gh[NB * 3 * HH];                 // 1536 floats
  const int dir = blockIdx.x;
  const float* gi  = dir ? gi1  : gi0;
  const float* whh = dir ? whh1 : whh0;
  const float* bhh = dir ? bhh1 : bhh0;
  const int tid = threadIdx.x;

  for (int e = tid; e < 384 * HH; e += 512) {       // transpose whh -> LDS
    const int c = e / HH, k = e % HH;
    whhT[k * 384 + c] = whh[e];
  }
  hcur[tid] = 0.f;
  __syncthreads();

  for (int t = 0; t < LSEQ; ++t) {
    const int tr = dir ? (LSEQ - 1 - t) : t;
    // phase A: gh[b][c] = bhh[c] + sum_k h[b][k] * whh[c][k]
#pragma unroll
    for (int s = 0; s < 3; ++s) {
      const int e = tid + 512 * s;
      const int b = e / 384, c = e % 384;
      float a = bhh[c];
      const float* hb = &hcur[b * HH];
#pragma unroll 8
      for (int k = 0; k < HH; ++k) a += hb[k] * whhT[k * 384 + c];
      gh[e] = a;
    }
    __syncthreads();
    // phase B: one GRU cell per thread (b = tid>>7, j = tid&127)
    const int b = tid >> 7, j = tid & 127;
    const size_t row = (size_t)tr * NB + b;
    const float* grow = gi + row * 384;
    const float ir = grow[j], iz = grow[HH + j], in = grow[2 * HH + j];
    const float hr = gh[b * 384 + j];
    const float hz = gh[b * 384 + HH + j];
    const float hn = gh[b * 384 + 2 * HH + j];
    const float rg = 1.f / (1.f + __expf(-(ir + hr)));
    const float zg = 1.f / (1.f + __expf(-(iz + hz)));
    const float ng = tanhf(in + rg * hn);
    const float hv = (1.f - zg) * ng + zg * hcur[tid];
    __syncthreads();
    hcur[tid] = hv;
    const size_t o = row * (2 * HH) + (size_t)dir * HH + j;
    xh16[o] = (_Float16)hv;
    if (out32) out32[o] = hv;
    __syncthreads();
  }
}

// ---------------------------------------------------------------------------
// host launcher
// ---------------------------------------------------------------------------
extern "C" void kernel_launch(void* const* d_in, const int* in_sizes, int n_in,
                              void* d_out, int out_size, void* d_ws, size_t ws_size,
                              hipStream_t stream) {
  (void)in_sizes; (void)n_in; (void)out_size; (void)ws_size;
  const float* passage = (const float*)d_in[0];
  const float* mask    = (const float*)d_in[1];
  const float* WvP     = (const float*)d_in[2];
  const float* WvPbar  = (const float*)d_in[3];
  const float* vT      = (const float*)d_in[4];
  const float* gateW   = (const float*)d_in[5];
  // gru params: d_in[6 + (layer*2+dir)*4 + {0:wih,1:whh,2:bih,3:bhh}]

  char* ws = (char*)d_ws;
  size_t off = 0;
  auto alloc = [&](size_t bytes) -> void* {
    off = (off + 255) & ~(size_t)255;
    void* p = ws + off;
    off += bytes;
    return p;
  };

  const int M = LSEQ * NB;                                   // 2048 rows
  _Float16* P16   = (_Float16*)alloc((size_t)M * DD * 2);    // passage f16
  _Float16* W1    = (_Float16*)alloc((size_t)HH * DD * 2);   // WvP f16
  _Float16* W2    = (_Float16*)alloc((size_t)HH * DD * 2);   // WvPbar f16
  _Float16* G16   = (_Float16*)alloc((size_t)2 * DD * 2 * DD * 2); // gate_w f16
  _Float16* WIH[6];
  for (int l = 0; l < 3; ++l)
    for (int d2 = 0; d2 < 2; ++d2) {
      const int k = (l == 0) ? 2 * DD : 2 * HH;
      WIH[l * 2 + d2] = (_Float16*)alloc((size_t)3 * HH * k * 2);
    }
  float*    ABUF  = (float*)alloc((size_t)M * HH * 4);
  float*    BBUF  = (float*)alloc((size_t)M * HH * 4);
  float*    CAT32 = (float*)alloc((size_t)M * 2 * DD * 4);
  _Float16* CAT16 = (_Float16*)alloc((size_t)M * 2 * DD * 2);
  _Float16* GATED = (_Float16*)alloc((size_t)M * 2 * DD * 2);
  float*    GI0   = (float*)alloc((size_t)M * 3 * HH * 4);
  float*    GI1   = (float*)alloc((size_t)M * 3 * HH * 4);
  _Float16* XH    = (_Float16*)alloc((size_t)M * 2 * HH * 2);

  auto cvt = [&](const float* s, _Float16* d, int n) {
    f32_to_f16_kernel<<<(n + 255) / 256, 256, 0, stream>>>(s, d, n);
  };

  // precision staging: f16 operands for all WMMA GEMMs
  cvt(passage, P16, M * DD);
  cvt(WvP,    W1, HH * DD);
  cvt(WvPbar, W2, HH * DD);
  cvt(gateW,  G16, 2 * DD * 2 * DD);
  for (int l = 0; l < 3; ++l)
    for (int d2 = 0; d2 < 2; ++d2) {
      const int k = (l == 0) ? 2 * DD : 2 * HH;
      cvt((const float*)d_in[6 + (l * 2 + d2) * 4 + 0], WIH[l * 2 + d2], 3 * HH * k);
    }

  // a = passage @ WvP^T ; b = passage @ WvPbar^T   (WMMA)
  wmma_gemm_kernel<0><<<(M / 16) * (HH / 16) / 4, 128, 0, stream>>>(
      P16, W1, ABUF, nullptr, nullptr, nullptr, M, HH, DD);
  wmma_gemm_kernel<0><<<(M / 16) * (HH / 16) / 4, 128, 0, stream>>>(
      P16, W2, BBUF, nullptr, nullptr, nullptr, M, HH, DD);

  // attention + softmax + context + concat
  attention_kernel<<<M, 256, 0, stream>>>(passage, mask, vT, ABUF, BBUF,
                                          CAT32, CAT16);

  // gated = cat * sigmoid(cat @ gate_w^T)   (WMMA, fused sigmoid-gate epilogue)
  wmma_gemm_kernel<2><<<(M / 16) * (2 * DD / 16) / 4, 128, 0, stream>>>(
      CAT16, G16, nullptr, nullptr, CAT32, GATED, M, 2 * DD, 2 * DD);

  // 3-layer bidirectional GRU
  const _Float16* xin = GATED;
  int Kin = 2 * DD;
  for (int l = 0; l < 3; ++l) {
    for (int d2 = 0; d2 < 2; ++d2) {
      const float* bih = (const float*)d_in[6 + (l * 2 + d2) * 4 + 2];
      float* gi = d2 ? GI1 : GI0;
      wmma_gemm_kernel<1><<<(M / 16) * (3 * HH / 16) / 4, 128, 0, stream>>>(
          xin, WIH[l * 2 + d2], gi, bih, nullptr, nullptr, M, 3 * HH, Kin);
    }
    const float* whh0 = (const float*)d_in[6 + (l * 2 + 0) * 4 + 1];
    const float* whh1 = (const float*)d_in[6 + (l * 2 + 1) * 4 + 1];
    const float* bhh0 = (const float*)d_in[6 + (l * 2 + 0) * 4 + 3];
    const float* bhh1 = (const float*)d_in[6 + (l * 2 + 1) * 4 + 3];
    float* o32 = (l == 2) ? (float*)d_out : nullptr;
    gru_scan_kernel<<<2, 512, (size_t)HH * 3 * HH * 4, stream>>>(
        GI0, GI1, whh0, whh1, bhh0, bhh1, XH, o32);
    xin = XH;
    Kin = 2 * HH;
  }
}
